// AttentionHead_57423712747731
// MI455X (gfx1250) — compile-verified, matrix operand order checked by reading, not run on hardware
//
#include <hip/hip_runtime.h>

// ---------------------------------------------------------------------------
// AttentionHead for MI455X (gfx1250, wave32, WMMA bf16 16x16x32, f32 accum)
//
//   q = (x @ Wq + bq) * 1/sqrt(128)      -> bf16 [B,L,128]
//   k =  y @ Wk + bk                     -> bf16 [B,L,128]
//   vT = (y @ Wv + bv)^T                 -> bf16 [B,128,L]   (K-major for WMMA B-frags)
//   out[q] = (1/Z) * sum_{key>q} exp(q.k) * v[key],  Z = sum_all exp(q.k)
//            (post-softmax triu mask, faithful to reference)
//
// Attention kernel stages K/V tiles into double-buffered LDS using CDNA5
// async global->LDS copies (ASYNCcnt), shared by all 8 waves of the block.
// ---------------------------------------------------------------------------

typedef __bf16 bf16;
typedef bf16  v8bf  __attribute__((ext_vector_type(8)));
typedef bf16  v16bf __attribute__((ext_vector_type(16)));
typedef float v8f   __attribute__((ext_vector_type(8)));
typedef int   v4i   __attribute__((ext_vector_type(4)));

typedef __attribute__((address_space(1))) v4i  gv4i;   // global int4
typedef __attribute__((address_space(3))) v4i  lv4i;   // LDS int4
typedef __attribute__((address_space(3))) char lchar;  // LDS byte

#define NB   4
#define SEQ  2048
#define XSZ  1024
#define PDM  128
#define KPAD 136   // 32-key K tile row stride (bf16): 272B = 68 dwords, conflict-free
#define VPAD 40    // V tile row stride (bf16): 80B = 20 dwords, conflict-free

__device__ __forceinline__ v8f wmma_bf16(v16bf a, v16bf b, v8f c) {
    return __builtin_amdgcn_wmma_f32_16x16x32_bf16(false, a, false, b,
                                                   (short)0, c, false, false);
}

// 16-bit A/B fragment (16x32 / 32x16), wave32 layout (ISA 7.12.2):
// lane: row = lane&15, half = lane>>4
//   VGPR0..3 hold K =      half*8 .. half*8+7   (contiguous)
//   VGPR4..7 hold K = 16 + half*8 .. half*8+7   (contiguous)
__device__ __forceinline__ v16bf load_frag(const bf16* rowptr, int half) {
    v8bf a = *(const v8bf*)(rowptr + half * 8);
    v8bf b = *(const v8bf*)(rowptr + 16 + half * 8);
    v16bf r;
#pragma unroll
    for (int i = 0; i < 8; ++i) { r[i] = a[i]; r[i + 8] = b[i]; }
    return r;
}

__device__ __forceinline__ v8f zero8() {
    v8f z = {0.f, 0.f, 0.f, 0.f, 0.f, 0.f, 0.f, 0.f};
    return z;
}

__device__ __forceinline__ float rowmax16(float v) {
    v = fmaxf(v, __shfl_xor(v, 1, 32));
    v = fmaxf(v, __shfl_xor(v, 2, 32));
    v = fmaxf(v, __shfl_xor(v, 4, 32));
    v = fmaxf(v, __shfl_xor(v, 8, 32));
    return v;
}
__device__ __forceinline__ float rowsum16(float v) {
    v += __shfl_xor(v, 1, 32);
    v += __shfl_xor(v, 2, 32);
    v += __shfl_xor(v, 4, 32);
    v += __shfl_xor(v, 8, 32);
    return v;
}

// CDNA5 async global->LDS copy of 16 bytes (GLOBAL_LOAD_ASYNC_TO_LDS_B128)
__device__ __forceinline__ void async_copy16(const bf16* gsrc, bf16* ldst) {
#if __has_builtin(__builtin_amdgcn_global_load_async_to_lds_b128)
    __builtin_amdgcn_global_load_async_to_lds_b128(
        (gv4i*)(unsigned long long)gsrc,     // int-to-AS(1) cast, no-op numerically
        (lv4i*)(lchar*)ldst,                 // addrspacecast generic->AS(3), then int4*
        0, 0);
#else
    unsigned loff = (unsigned)(unsigned long long)(lchar*)ldst;
    unsigned long long ga = (unsigned long long)gsrc;
    asm volatile("global_load_async_to_lds_b128 %0, %1, off"
                 :: "v"(loff), "v"(ga) : "memory");
#endif
}

__device__ __forceinline__ void wait_async0() {
#if __has_builtin(__builtin_amdgcn_s_wait_asynccnt)
    __builtin_amdgcn_s_wait_asynccnt(0);
#else
    asm volatile("s_wait_asynccnt 0" ::: "memory");
#endif
}

// ---------------------------------------------------------------------------
// Kernel 0: transpose + convert weights: Wt[d][kk] = bf16(W[kk][d]), 128x1024
// ---------------------------------------------------------------------------
__global__ __launch_bounds__(256) void wt_kernel(const float* __restrict__ Wq,
                                                 const float* __restrict__ Wk,
                                                 const float* __restrict__ Wv,
                                                 bf16* __restrict__ wtq,
                                                 bf16* __restrict__ wtk,
                                                 bf16* __restrict__ wtv) {
    int idx = blockIdx.x * 256 + threadIdx.x;         // 3 * 1024 * 128 total
    int mat = idx / (XSZ * PDM);
    int rem = idx % (XSZ * PDM);
    int d  = rem / XSZ;
    int kk = rem % XSZ;
    const float* W = (mat == 0) ? Wq : (mat == 1) ? Wk : Wv;
    bf16*       Wt = (mat == 0) ? wtq : (mat == 1) ? wtk : wtv;
    Wt[(size_t)d * XSZ + kk] = (bf16)W[(size_t)kk * PDM + d];
}

// ---------------------------------------------------------------------------
// Kernel 1: fused QKV projection. One wave = one 16-row tile of one matrix.
// ---------------------------------------------------------------------------
__global__ __launch_bounds__(256) void proj_kernel(const float* __restrict__ x,
                                                   const float* __restrict__ y,
                                                   const float* __restrict__ bq,
                                                   const float* __restrict__ bk,
                                                   const float* __restrict__ bv,
                                                   const bf16* __restrict__ wtq,
                                                   const bf16* __restrict__ wtk,
                                                   const bf16* __restrict__ wtv,
                                                   bf16* __restrict__ qb,
                                                   bf16* __restrict__ kb,
                                                   bf16* __restrict__ vtb) {
    const int wid  = threadIdx.x >> 5;
    const int lane = threadIdx.x & 31;
    const int task = blockIdx.x * 8 + wid;            // 0 .. 3*512-1
    const int mat  = task / 512;                      // 0=q 1=k 2=v
    const int rt   = task % 512;
    const int b    = rt >> 7;
    const int rowbase = (rt & 127) * 16;
    const int N    = lane & 15;
    const int half = lane >> 4;

    const float* src  = ((mat == 0) ? x : y) + ((size_t)b * SEQ + rowbase + N) * XSZ;
    const bf16*  Wt   = (mat == 0) ? wtq : (mat == 1) ? wtk : wtv;
    const float* bias = (mat == 0) ? bq : (mat == 1) ? bk : bv;

    v8f acc[8];
#pragma unroll
    for (int j = 0; j < 8; ++j) acc[j] = zero8();

    for (int kc = 0; kc < XSZ; kc += 32) {
        v8f a0 = *(const v8f*)(src + kc + half * 8);
        v8f a1 = *(const v8f*)(src + kc + 16 + half * 8);
        v16bf af;
#pragma unroll
        for (int i = 0; i < 8; ++i) { af[i] = (bf16)a0[i]; af[i + 8] = (bf16)a1[i]; }
#pragma unroll
        for (int j = 0; j < 8; ++j) {
            v16bf bfb = load_frag(Wt + (size_t)(j * 16 + N) * XSZ + kc, half);
            acc[j] = wmma_bf16(af, bfb, acc[j]);
        }
    }

    const float scale = 0.08838834764831845f;         // 1/sqrt(128)
#pragma unroll
    for (int j = 0; j < 8; ++j) {
        int col = j * 16 + N;
        float bb = bias[col];
#pragma unroll
        for (int r = 0; r < 8; ++r) {
            int row = rowbase + r + 8 * half;
            float v = acc[j][r] + bb;
            if (mat == 0)      qb[((size_t)b * SEQ + row) * PDM + col] = (bf16)(v * scale);
            else if (mat == 1) kb[((size_t)b * SEQ + row) * PDM + col] = (bf16)v;
            else               vtb[((size_t)b * PDM + col) * SEQ + row] = (bf16)v;
        }
    }
}

// ---------------------------------------------------------------------------
// Kernel 2: fused flash-style attention, 16 q rows per wave, 32 keys/step.
// K/V tiles staged once per block via async global->LDS, double-buffered.
// ---------------------------------------------------------------------------
__global__ __launch_bounds__(256) void attn_kernel(const bf16* __restrict__ qb,
                                                   const bf16* __restrict__ kb,
                                                   const bf16* __restrict__ vtb,
                                                   const int* __restrict__ maskp,
                                                   float* __restrict__ out) {
    __shared__ bf16 Klds[2][32][KPAD];    // 2 x 8.5KB : K rows (K-major, padded)
    __shared__ bf16 Vlds[2][PDM][VPAD];   // 2 x 10KB  : vT rows (padded)
    __shared__ bf16 Pbuf[8][16][32];      // per-wave P staging, 8KB

    const int tid  = threadIdx.x;
    const int wid  = tid >> 5;
    const int lane = tid & 31;
    const int gw   = blockIdx.x * 8 + wid;            // 0 .. 4*128-1
    const int b    = gw / (SEQ / 16);                 // uniform per block (128%8==0)
    const int qbase = (gw % (SEQ / 16)) * 16;
    const int N    = lane & 15;
    const int half = lane >> 4;
    const int msk  = *maskp;

    const bf16* kbase_g = kb  + (size_t)b * SEQ * PDM;
    const bf16* vbase_g = vtb + (size_t)b * PDM * SEQ;

    // cooperative async stage of one 32-key K tile + V tile (4 x b128 per thread)
    auto issue_stage = [&](int st, int kbi) {
#pragma unroll
        for (int jj = 0; jj < 4; ++jj) {
            int o = tid + 256 * jj;                   // 0..1023
            if (o < 512) {                            // K: 32 rows x 256B
                int row = o >> 4, seg = o & 15;
                async_copy16(kbase_g + (size_t)(kbi + row) * PDM + seg * 8,
                             &Klds[st][row][seg * 8]);
            } else {                                  // V: 128 rows x 64B
                int o2 = o - 512;
                int d = o2 >> 2, seg = o2 & 3;
                async_copy16(vbase_g + (size_t)d * SEQ + kbi + seg * 8,
                             &Vlds[st][d][seg * 8]);
            }
        }
    };

    // Q fragments for the whole head dim (scale folded into q)
    const bf16* qrow = qb + ((size_t)b * SEQ + qbase + N) * PDM;
    v16bf qf[4];
#pragma unroll
    for (int c = 0; c < 4; ++c) qf[c] = load_frag(qrow + c * 32, half);

    float m[8], l[8];
    v8f acc[8];
#pragma unroll
    for (int r = 0; r < 8; ++r) { m[r] = -__builtin_inff(); l[r] = 0.f; }
#pragma unroll
    for (int j = 0; j < 8; ++j) acc[j] = zero8();

    issue_stage(0, 0);

    for (int it = 0; it < SEQ / 32; ++it) {
        const int st  = it & 1;
        const int kbi = it * 32;

        wait_async0();                  // my stage-st copies have landed in LDS
        __syncthreads();                // everyone's stage-st copies visible
        if (kbi + 32 < SEQ) issue_stage(st ^ 1, kbi + 32);   // overlap next stage

        // ---- S = Q @ K^T for 32 keys (two 16x16 tiles), from LDS ----
        v8f S0 = zero8(), S1 = zero8();
#pragma unroll
        for (int c = 0; c < 4; ++c) {
            v16bf kf0 = load_frag(&Klds[st][N][c * 32], half);
            S0 = wmma_bf16(qf[c], kf0, S0);
        }
#pragma unroll
        for (int c = 0; c < 4; ++c) {
            v16bf kf1 = load_frag(&Klds[st][16 + N][c * 32], half);
            S1 = wmma_bf16(qf[c], kf1, S1);
        }

        // ---- online softmax over ALL keys (mask is post-softmax) ----
        float mnew[8], cf[8];
#pragma unroll
        for (int r = 0; r < 8; ++r) {
            float bm = rowmax16(fmaxf(S0[r], S1[r]));
            mnew[r] = fmaxf(m[r], bm);
            cf[r]   = __expf(m[r] - mnew[r]);
            m[r]    = mnew[r];
        }
#pragma unroll
        for (int r = 0; r < 8; ++r) {
            float e0 = __expf(S0[r] - mnew[r]);
            float e1 = __expf(S1[r] - mnew[r]);
            l[r] = l[r] * cf[r] + rowsum16(e0 + e1);
            // triu(k=1): keep only keys strictly greater than the query index
            int qrow_g = qbase + r + 8 * half;
            bf16 w0 = (bf16)((!msk || (kbi + N)      > qrow_g) ? e0 : 0.f);
            bf16 w1 = (bf16)((!msk || (kbi + 16 + N) > qrow_g) ? e1 : 0.f);
            Pbuf[wid][r + 8 * half][N]      = w0;
            Pbuf[wid][r + 8 * half][16 + N] = w1;
        }
#pragma unroll
        for (int j = 0; j < 8; ++j)
#pragma unroll
            for (int r = 0; r < 8; ++r) acc[j][r] *= cf[r];

        asm volatile("s_wait_dscnt 0" ::: "memory");   // same-wave P relayout

        // ---- acc += P(16x32) @ V(32x128), V from LDS ----
        v16bf pf = load_frag(&Pbuf[wid][N][0], half);
#pragma unroll
        for (int j = 0; j < 8; ++j) {
            v16bf vf = load_frag(&Vlds[st][j * 16 + N][0], half);
            acc[j] = wmma_bf16(pf, vf, acc[j]);
        }

        __syncthreads();                // all reads of stage st done before reuse
    }

    // ---- epilogue: out = acc / Z ----
#pragma unroll
    for (int j = 0; j < 8; ++j) {
#pragma unroll
        for (int r = 0; r < 8; ++r) {
            int row = qbase + r + 8 * half;
            int col = j * 16 + N;
            out[((size_t)b * SEQ + row) * PDM + col] = acc[j][r] / l[r];
        }
    }
}

// ---------------------------------------------------------------------------
// Launch. Workspace layout (needs 6.75 MB):
//   [0,2MB)   q  bf16 [4,2048,128]
//   [2,4MB)   k  bf16 [4,2048,128]
//   [4,6MB)   vT bf16 [4,128,2048]
//   [6MB,..)  WqT/WkT/WvT bf16 [128,1024] (256KB each)
// ---------------------------------------------------------------------------
extern "C" void kernel_launch(void* const* d_in, const int* in_sizes, int n_in,
                              void* d_out, int out_size, void* d_ws, size_t ws_size,
                              hipStream_t stream) {
    const float* x  = (const float*)d_in[0];
    const float* y  = (const float*)d_in[1];
    const float* Wq = (const float*)d_in[2];
    const float* bq = (const float*)d_in[3];
    const float* Wk = (const float*)d_in[4];
    const float* bk = (const float*)d_in[5];
    const float* Wv = (const float*)d_in[6];
    const float* bv = (const float*)d_in[7];
    const int* mask = (const int*)d_in[8];
    float* out = (float*)d_out;

    char* ws = (char*)d_ws;
    bf16* qb  = (bf16*)(ws);
    bf16* kb  = (bf16*)(ws + (size_t)2 * 1024 * 1024);
    bf16* vtb = (bf16*)(ws + (size_t)4 * 1024 * 1024);
    bf16* wtq = (bf16*)(ws + (size_t)6 * 1024 * 1024);
    bf16* wtk = (bf16*)(ws + (size_t)6 * 1024 * 1024 + 256 * 1024);
    bf16* wtv = (bf16*)(ws + (size_t)6 * 1024 * 1024 + 512 * 1024);

    wt_kernel<<<dim3(1536), dim3(256), 0, stream>>>(Wq, Wk, Wv, wtq, wtk, wtv);
    proj_kernel<<<dim3(192), dim3(256), 0, stream>>>(x, y, bq, bk, bv,
                                                     wtq, wtk, wtv, qb, kb, vtb);
    attn_kernel<<<dim3(64), dim3(256), 0, stream>>>(qb, kb, vtb, mask, out);
}
